// RNNV1_86844238725316
// MI455X (gfx1250) — compile-verified, matrix operand order checked by reading, not run on hardware
//
#include <hip/hip_runtime.h>
#include <hip/hip_bf16.h>

// ---- CDNA5 WMMA / descriptor vector types -------------------------------
typedef __attribute__((ext_vector_type(16))) _Float16     v16h;
typedef __attribute__((ext_vector_type(8)))  _Float16     v8h;
typedef __attribute__((ext_vector_type(8)))  float        v8f;
typedef __attribute__((ext_vector_type(4)))  unsigned int u32x4;
typedef __attribute__((ext_vector_type(8)))  unsigned int u32x8;

#define VOCAB 32000
#define EMB   256
#define IN    4096
#define HID   16
#define NL    5
#define TT    128
#define NE    8
#define G4    64          // 4*HID gates

__device__ __forceinline__ float sigf(float x) { return 1.0f / (1.0f + __expf(-x)); }

__device__ __forceinline__ v16h cat16(v8h lo, v8h hi) {
    return __builtin_shufflevector(lo, hi, 0,1,2,3,4,5,6,7,8,9,10,11,12,13,14,15);
}

// LDS byte offset of a __shared__ object: generic LDS addresses are
// {aperture_hi32 | lds_offset_lo32} on AMDGPU, so truncation yields the offset.
__device__ __forceinline__ unsigned lds_off(const void* p) {
    return (unsigned)(unsigned long long)p;
}

// ---- Kernel 1: W_ih0 (64x4096 f32) -> f16, same layout ------------------
__global__ void convert_w(const float* __restrict__ W, _Float16* __restrict__ Wf16) {
    int i = blockIdx.x * blockDim.x + threadIdx.x;
    if (i < G4 * IN) Wf16[i] = (_Float16)W[i];
}

// ---- Kernel 2: concat recurrent weights per layer: Wcat[l][g][k0..31] ---
__global__ void build_wcat(const float* __restrict__ W_ihr,
                           const float* __restrict__ W_hh,
                           _Float16* __restrict__ Wcat) {
    int i = blockIdx.x * blockDim.x + threadIdx.x;
    if (i >= NL * G4 * 32) return;
    int l = i / (G4 * 32);
    int rem = i % (G4 * 32);
    int g = rem / 32, k = rem % 32;
    float v;
    if (k < HID) v = (l == 0) ? 0.0f : W_ihr[((l - 1) * G4 + g) * HID + k];
    else         v = W_hh[(l * G4 + g) * HID + (k - HID)];
    Wcat[i] = (_Float16)v;
}

// ---- Kernel 3: hoisted layer-0 input projection -------------------------
// proj[t][e][g] = sum_k emb[da[t][k]][e] * W_ih0[g][k]
// grid (T=128, e-tiles=16); 128 threads = 4 waves; wave w owns gate tile g0=16w.
// B tiles (32K x 64G f16, row stride 4096) are fetched by the Tensor Data
// Mover into LDS; the VALU/VMEM pipes handle only the embedding gather.
__global__ __launch_bounds__(128)
void proj_gemm(const float* __restrict__ emb, const int* __restrict__ da,
               const _Float16* __restrict__ Wf16, float* __restrict__ proj) {
    __shared__ __align__(16) _Float16 Asub[16][32];   // [m][k] tile, 1KB
    __shared__ __align__(16) _Float16 Bsub[64][32];   // [g][k] tile, 4KB (TDM dest)

    const int t   = blockIdx.x;
    const int e0  = blockIdx.y * 16;
    const int tid = threadIdx.x;
    const int w = tid >> 5, L = tid & 31, mn = L & 15, s = L >> 4;

    const int kq = tid >> 2;   // gather row within chunk (0..31)
    const int q  = tid & 3;    // quarter of 16 e-values

    const unsigned bsub_lds = lds_off(&Bsub[0][0]);
    const unsigned long long wbase = (unsigned long long)Wf16;

    v8f acc = {};
    for (int kc = 0; kc < IN / 32; ++kc) {
        const int k0 = kc * 32;
        // ---- TDM: DMA the 32x64 f16 weight tile into LDS (wave 0 only) ----
        if (w == 0) {
            unsigned long long ga = wbase + (unsigned long long)k0 * 2u;
            u32x4 g0;
            g0.x = 1u;                                   // count=1 (valid D#)
            g0.y = bsub_lds;                             // lds_addr
            g0.z = (unsigned)ga;                         // global_addr[31:0]
            g0.w = (unsigned)(ga >> 32) | 0x80000000u;   // addr[56:32] | type=2
            u32x8 g1;
            g1.s0 = 0x00010000u;          // wg_mask=0, data_size=1 (2 bytes)
            g1.s1 = (IN & 0xFFFFu) << 16; // tensor_dim0 = 4096 (bits 79:48)
            g1.s2 = (G4 & 0xFFFFu) << 16; // tensor_dim1 = 64   (bits 111:80)
            g1.s3 = 32u << 16;            // tile_dim0 = 32     (bits 127:112)
            g1.s4 = G4;                   // tile_dim1 = 64, tile_dim2 = 0
            g1.s5 = IN;                   // tensor_dim0_stride = 4096
            g1.s6 = 0u;                   // dim1_stride (unused for 2D tile)
            g1.s7 = 0u;
            asm volatile("tensor_load_to_lds %0, %1"
                         :: "s"(g0), "s"(g1) : "memory");
            __builtin_amdgcn_s_wait_tensorcnt(0);
        }
        // ---- gather A tile: column k = emb row da[t][k0+k], elems e0..e0+15
        int row = da[t * IN + k0 + kq];
        float4 v = *(const float4*)(emb + (size_t)row * EMB + e0 + q * 4);
        Asub[q * 4 + 0][kq] = (_Float16)v.x;
        Asub[q * 4 + 1][kq] = (_Float16)v.y;
        Asub[q * 4 + 2][kq] = (_Float16)v.z;
        Asub[q * 4 + 3][kq] = (_Float16)v.w;
        // prefetch next chunk's embedding rows (global_prefetch_b8)
        if (kc + 1 < IN / 32) {
            int nrow = da[t * IN + k0 + 32 + kq];
            __builtin_prefetch(emb + (size_t)nrow * EMB + e0, 0, 1);
        }
        __syncthreads();
        // A fragment: lane L (m=L&15), halves 0..7 -> K=8s.., 8..15 -> K=16+8s..
        v8h alo = *(const v8h*)&Asub[mn][8 * s];
        v8h ahi = *(const v8h*)&Asub[mn][16 + 8 * s];
        v8h blo = *(const v8h*)&Bsub[16 * w + mn][8 * s];
        v8h bhi = *(const v8h*)&Bsub[16 * w + mn][16 + 8 * s];
        acc = __builtin_amdgcn_wmma_f32_16x16x32_f16(
                  false, cat16(alo, ahi), false, cat16(blo, bhi),
                  (short)0, acc, false, false);
        __syncthreads();
    }
    // D layout: (vgpr r, lane L) -> M = r + 8*(L>>4), N = L&15
#pragma unroll
    for (int r = 0; r < 8; ++r) {
        int e = e0 + r + 8 * s;
        int g = 16 * w + mn;
        proj[(size_t)t * (EMB * G4) + e * G4 + g] = acc[r];
    }
}

// ---- Kernel 4: sequential LSTM recurrence + fused output linear ---------
// One persistent workgroup (256 threads = 8 waves). h state lives in LDS as
// f16 (directly usable as WMMA A rows), c state lives in registers. Per-layer
// weights are pulled into LDS with async global->LDS copies (ASYNCcnt).
__global__ __launch_bounds__(256)
void lstm_seq(const float* __restrict__ proj, const _Float16* __restrict__ Wcat,
              const float* __restrict__ bias, const float* __restrict__ W_lin,
              const float* __restrict__ b_lin, float* __restrict__ out) {
    __shared__ __align__(16) float    gates[EMB][G4];       // 64KB
    __shared__ __align__(16) _Float16 hbuf[NL][EMB][HID];   // 40KB
    __shared__ __align__(16) _Float16 Bl[G4][32];           // 4KB (async dest)

    const int tid = threadIdx.x;
    const int w = tid >> 5, L = tid & 31, mn = L & 15, s = L >> 4;
    // async-copy assignment: thread tid moves 8 halves (16B)
    const int ag = tid >> 2, aq = tid & 3;
    const unsigned bl_dst = lds_off(&Bl[ag][aq * 8]);

    float creg[NL][HID];
#pragma unroll
    for (int l = 0; l < NL; ++l)
#pragma unroll
        for (int j = 0; j < HID; ++j) {
            creg[l][j] = 0.0f;
            hbuf[l][tid][j] = (_Float16)0.0f;
        }
    __syncthreads();

    for (int t = 0; t < TT; ++t) {
#pragma unroll
        for (int l = 0; l < NL; ++l) {
            // stage this layer's [W_ih | W_hh] (K=32, g-major) asynchronously
            {
                const _Float16* src = Wcat + ((l * G4 + ag) * 32 + aq * 8);
                asm volatile("global_load_async_to_lds_b128 %0, %1, off"
                             :: "v"(bl_dst), "v"(src) : "memory");
                asm volatile("s_wait_asynccnt 0x0" ::: "memory");
            }
            __syncthreads();
            // 64 (e-tile, g-tile) WMMA tiles over 8 waves: one 16x16x32 each.
            // A row e = [ inp(e)[0..15] | h_l(e)[0..15] ]
#pragma unroll
            for (int it = 0; it < 8; ++it) {
                int tile = w * 8 + it;
                int et = tile >> 2, gt = tile & 3;
                int erow = et * 16 + mn;
                v8h alo = {};
                if (l != 0) alo = *(const v8h*)&hbuf[l - 1][erow][8 * s];
                v8h ahi = *(const v8h*)&hbuf[l][erow][8 * s];
                v8h blo = *(const v8h*)&Bl[gt * 16 + mn][8 * s];
                v8h bhi = *(const v8h*)&Bl[gt * 16 + mn][16 + 8 * s];
                // bias (+ precomputed input projection for layer 0) via C operand
                v8f cin;
#pragma unroll
                for (int r = 0; r < 8; ++r) {
                    int e = et * 16 + r + 8 * s;
                    int g = gt * 16 + mn;
                    float base = bias[l * G4 + g];
                    if (l == 0) base += proj[(size_t)t * (EMB * G4) + e * G4 + g];
                    cin[r] = base;
                }
                v8f d = __builtin_amdgcn_wmma_f32_16x16x32_f16(
                            false, cat16(alo, ahi), false, cat16(blo, bhi),
                            (short)0, cin, false, false);
#pragma unroll
                for (int r = 0; r < 8; ++r) {
                    int e = et * 16 + r + 8 * s;
                    int g = gt * 16 + mn;
                    gates[e][g] = d[r];
                }
            }
            __syncthreads();
            // LSTM cell (PyTorch gate order i,f,g,o); thread tid owns row e=tid
#pragma unroll
            for (int j = 0; j < HID; ++j) {
                float ig = gates[tid][j];
                float fg = gates[tid][HID + j];
                float gg = gates[tid][2 * HID + j];
                float og = gates[tid][3 * HID + j];
                float cn = sigf(fg) * creg[l][j] + sigf(ig) * tanhf(gg);
                creg[l][j] = cn;
                hbuf[l][tid][j] = (_Float16)(sigf(og) * tanhf(cn));
            }
            __syncthreads();
        }
    }
    // out[ne] = relu(h[4]).flatten() @ W_lin[ne] + b_lin[ne]
    float* gflat = &gates[0][0];
#pragma unroll
    for (int ne = 0; ne < NE; ++ne) {
        float p = 0.0f;
#pragma unroll
        for (int j = 0; j < HID; ++j) {
            float hv = (float)hbuf[NL - 1][tid][j];
            hv = hv > 0.0f ? hv : 0.0f;
            p += hv * W_lin[ne * IN + tid * HID + j];
        }
        gflat[ne * EMB + tid] = p;
    }
    __syncthreads();
    if (tid < NE) {
        float ssum = 0.0f;
        for (int q = 0; q < EMB; ++q) ssum += gflat[tid * EMB + q];
        out[tid] = ssum + b_lin[tid];
    }
}

// ---- Host-side launch ----------------------------------------------------
extern "C" void kernel_launch(void* const* d_in, const int* in_sizes, int n_in,
                              void* d_out, int out_size, void* d_ws, size_t ws_size,
                              hipStream_t stream) {
    const float* emb   = (const float*)d_in[0];
    const float* W_ih0 = (const float*)d_in[1];
    const float* W_ihr = (const float*)d_in[2];
    const float* W_hh  = (const float*)d_in[3];
    const float* bias  = (const float*)d_in[4];
    const float* W_lin = (const float*)d_in[5];
    const float* b_lin = (const float*)d_in[6];
    const int*   da    = (const int*)d_in[7];
    float* out = (float*)d_out;

    char* ws = (char*)d_ws;
    _Float16* Wf16 = (_Float16*)ws;                         // 512KB: f16 W_ih0
    _Float16* Wcat = (_Float16*)(ws + (size_t)G4 * IN * 2); // 20KB: recurrent weights
    float*    proj = (float*)(ws + (1u << 20));             // 8MB:  layer-0 projections

    convert_w<<<(G4 * IN + 255) / 256, 256, 0, stream>>>(W_ih0, Wf16);
    build_wcat<<<(NL * G4 * 32 + 255) / 256, 256, 0, stream>>>(W_ihr, W_hh, Wcat);

    dim3 g2(TT, EMB / 16);
    proj_gemm<<<g2, 128, 0, stream>>>(emb, da, Wf16, proj);

    lstm_seq<<<1, 256, 0, stream>>>(proj, Wcat, bias, W_lin, b_lin, out);
}
